// SetGNN_87016037417181
// MI455X (gfx1250) — compile-verified, matrix operand order checked by reading, not run on hardware
//
#include <hip/hip_runtime.h>
#include <hip/hip_bf16.h>

typedef __attribute__((ext_vector_type(16))) _Float16 v16h;
typedef __attribute__((ext_vector_type(8)))  float    v8f;

#define D_MODEL   128
#define N_NODES_C 100000
#define N_HE_C    20000
#define WFRAG     16384   // halfs per packed 128x128 weight matrix (8 tiles * 4 chunks * 32 lanes * 16)

union FragH {
    v16h     v;
    unsigned u[8];
    uint4    q[2];
    _Float16 h[16];
};

// ---------------------------------------------------------------------------
// Pack a 128x128 f32 weight matrix (row-major, W[k][n]) into per-lane WMMA
// B-fragments (f16).  Fragment (t = N-tile 0..7, c = K-chunk 0..3) for lane L:
//   N = t*16 + (L%16), half = L/16
//   element e (v=e/2, p=e%2, g=v/4, vv=v%4): K = 32c + 16g + 8*half + 2vv + p
// Stored contiguously: out[((t*4+c)*32 + lane)*16 + e]
// ---------------------------------------------------------------------------
__global__ void pack_weights_kernel(const float* __restrict__ W,
                                    _Float16* __restrict__ out) {
    const int tid = blockIdx.x * blockDim.x + threadIdx.x;
    if (tid >= 1024) return;
    const int lane = tid & 31;
    const int c    = (tid >> 5) & 3;
    const int t    = tid >> 7;
    const int n    = t * 16 + (lane & 15);
    const int hf   = lane >> 4;
    FragH f;
#pragma unroll
    for (int e = 0; e < 16; ++e) {
        const int v = e >> 1, p = e & 1, g = v >> 2, vv = v & 3;
        const int k = 32 * c + 16 * g + 8 * hf + 2 * vv + p;
        f.h[e] = (_Float16)W[k * D_MODEL + n];
    }
    uint4* dst = reinterpret_cast<uint4*>(out) + (size_t)tid * 2;
    dst[0] = f.q[0];
    dst[1] = f.q[1];
}

// ---------------------------------------------------------------------------
// One 16x128 @ 128x128 GEMM for one wave: A-fragments from LDS (f16 row-major
// 16x128 tile), B-fragments from packed global weights. acc[t] = 16x16 f32.
// ---------------------------------------------------------------------------
__device__ __forceinline__ void wmma_gemm_128(const _Float16* __restrict__ As,
                                              const _Float16* __restrict__ Wp,
                                              int lane, v8f acc[8]) {
    const int m  = lane & 15;
    const int hf = lane >> 4;
#pragma unroll
    for (int c = 0; c < 4; ++c) {
        FragH a;
#pragma unroll
        for (int v = 0; v < 8; ++v) {
            const int g = v >> 2, vv = v & 3;
            const int k = 32 * c + 16 * g + 8 * hf + 2 * vv;
            a.u[v] = *reinterpret_cast<const unsigned*>(As + m * D_MODEL + k);
        }
#pragma unroll
        for (int t = 0; t < 8; ++t) {
            FragH b;
            const uint4* bp =
                reinterpret_cast<const uint4*>(Wp) + (size_t)((t * 4 + c) * 32 + lane) * 2;
            b.q[0] = bp[0];
            b.q[1] = bp[1];
            acc[t] = __builtin_amdgcn_wmma_f32_16x16x32_f16(
                false, a.v, false, b.v, (short)0, acc[t], false, false);
        }
    }
}

// ---------------------------------------------------------------------------
// Fused SetConv MLP kernel. One wave = one 16-row stripe; 4 waves / block.
//  has_wc=1 (encoder): relu(LN(relu(X@W1+b1))@W2+b2) @ Wc        -> Out (f32)
//  has_wc=0 (decoder): relu(LN(relu(X@W1+b1))@W2+b2)             -> Out (f32)
// ---------------------------------------------------------------------------
__global__ __launch_bounds__(128) void setconv_mlp_kernel(
    const float* __restrict__ X, float* __restrict__ Out, int n,
    const _Float16* __restrict__ Wp1, const _Float16* __restrict__ Wp2,
    const _Float16* __restrict__ Wpc,
    const float* __restrict__ b1v, const float* __restrict__ gv,
    const float* __restrict__ bev, const float* __restrict__ b2v, int has_wc) {
    __shared__ _Float16 Abuf[4][16 * D_MODEL];
    __shared__ float    Hbuf[4][16 * D_MODEL];

    const int  wave   = threadIdx.x >> 5;
    const int  lane   = threadIdx.x & 31;
    const int  row0   = (blockIdx.x * 4 + wave) * 16;
    const bool active = row0 < n;   // n is a multiple of 16 -> whole stripes only
    const int  m      = lane & 15;
    const int  hf     = lane >> 4;
    _Float16* As = Abuf[wave];
    float*    Hs = Hbuf[wave];

    // stage X (f32) -> As (f16), 16 rows x 128 cols
    if (active) {
        for (int idx = lane; idx < 512; idx += 32) {
            const int pos = idx << 2;
            const int r   = pos >> 7;
            const int k   = pos & 127;
            const float4 xv =
                *reinterpret_cast<const float4*>(X + (size_t)(row0 + r) * D_MODEL + k);
            _Float16* dst = As + r * D_MODEL + k;
            dst[0] = (_Float16)xv.x; dst[1] = (_Float16)xv.y;
            dst[2] = (_Float16)xv.z; dst[3] = (_Float16)xv.w;
        }
    }
    __syncthreads();

    v8f acc[8];
    const v8f zacc = {};

    // ---- GEMM1: H = relu(X@W1 + b1) -> Hs (f32) ----
    if (active) {
#pragma unroll
        for (int t = 0; t < 8; ++t) acc[t] = zacc;
        wmma_gemm_128(As, Wp1, lane, acc);
#pragma unroll
        for (int t = 0; t < 8; ++t) {
            const int N = t * 16 + m;
#pragma unroll
            for (int r = 0; r < 8; ++r) {
                const int M = r + 8 * hf;
                Hs[M * D_MODEL + N] = fmaxf(acc[t][r] + b1v[N], 0.0f);
            }
        }
    }
    __syncthreads();

    // ---- LayerNorm over each row (2 lanes per row, 64 cols each) -> As (f16)
    if (active) {
        const int row   = m;
        const int cbase = hf * 64;
        float s = 0.0f, s2 = 0.0f;
        for (int k = 0; k < 64; ++k) {
            const float h = Hs[row * D_MODEL + cbase + k];
            s += h; s2 += h * h;
        }
        s  += __shfl_xor(s, 16, 32);
        s2 += __shfl_xor(s2, 16, 32);
        const float mean = s * (1.0f / 128.0f);
        float var = s2 * (1.0f / 128.0f) - mean * mean;
        var = fmaxf(var, 0.0f);
        const float rstd = rsqrtf(var + 1e-5f);
        for (int k = 0; k < 64; ++k) {
            const int col = cbase + k;
            const float h = Hs[row * D_MODEL + col];
            As[row * D_MODEL + col] = (_Float16)((h - mean) * rstd * gv[col] + bev[col]);
        }
    }
    __syncthreads();

    // ---- GEMM2: Y = LN(H)@W2 + b2, relu ----
    if (active) {
#pragma unroll
        for (int t = 0; t < 8; ++t) acc[t] = zacc;
        wmma_gemm_128(As, Wp2, lane, acc);
        if (has_wc) {
#pragma unroll
            for (int t = 0; t < 8; ++t) {
                const int N = t * 16 + m;
#pragma unroll
                for (int r = 0; r < 8; ++r) {
                    const int M = r + 8 * hf;
                    As[M * D_MODEL + N] = (_Float16)fmaxf(acc[t][r] + b2v[N], 0.0f);
                }
            }
        } else {
#pragma unroll
            for (int t = 0; t < 8; ++t) {
                const int N = t * 16 + m;
#pragma unroll
                for (int r = 0; r < 8; ++r) {
                    const int M = r + 8 * hf;
                    Out[(size_t)(row0 + M) * D_MODEL + N] = fmaxf(acc[t][r] + b2v[N], 0.0f);
                }
            }
        }
    }
    __syncthreads();

    // ---- GEMM3 (encoder only): XW = Y @ Wc -> Out (f32, no relu/bias) ----
    if (active && has_wc) {
#pragma unroll
        for (int t = 0; t < 8; ++t) acc[t] = zacc;
        wmma_gemm_128(As, Wpc, lane, acc);
#pragma unroll
        for (int t = 0; t < 8; ++t) {
            const int N = t * 16 + m;
#pragma unroll
            for (int r = 0; r < 8; ++r) {
                const int M = r + 8 * hf;
                Out[(size_t)(row0 + M) * D_MODEL + N] = acc[t][r];
            }
        }
    }
}

// ---------------------------------------------------------------------------
// minimum of hyperedge ids (reference shifts hedges by their min)
// ---------------------------------------------------------------------------
__global__ void min_reduce_kernel(const int* __restrict__ idx, int n,
                                  int* __restrict__ out) {
    const int tid    = blockIdx.x * blockDim.x + threadIdx.x;
    const int stride = gridDim.x * blockDim.x;
    int local = 0x7fffffff;
    for (int i = tid; i < n; i += stride) local = min(local, idx[i]);
    atomicMin(out, local);
}

// ---------------------------------------------------------------------------
// segment-sum scatter: one wave per incidence row, float4 per lane (128 dims)
// ---------------------------------------------------------------------------
__global__ void scatter_add_kernel(const float* __restrict__ XW,
                                   const int* __restrict__ srcIdx,
                                   const int* __restrict__ tgtIdx,
                                   const int* __restrict__ minp,
                                   int sub_src, int sub_tgt,
                                   float* __restrict__ agg,
                                   float* __restrict__ deg, int n_inc) {
    const long long tid = (long long)blockIdx.x * blockDim.x + threadIdx.x;
    const int inc = (int)(tid >> 5);
    if (inc >= n_inc) return;
    const int lp = (int)(tid & 31);
    const int mn = *minp;
    const int s  = srcIdx[inc] - (sub_src ? mn : 0);
    const int tg = tgtIdx[inc] - (sub_tgt ? mn : 0);
    const float4 v = *reinterpret_cast<const float4*>(XW + (size_t)s * D_MODEL + lp * 4);
    float* ap = agg + (size_t)tg * D_MODEL + lp * 4;
    atomicAdd(ap + 0, v.x);
    atomicAdd(ap + 1, v.y);
    atomicAdd(ap + 2, v.z);
    atomicAdd(ap + 3, v.w);
    if (lp == 0) atomicAdd(deg + tg, 1.0f);
}

// agg /= max(deg, 1)
__global__ void seg_norm_kernel(float* __restrict__ agg,
                                const float* __restrict__ deg, int nseg) {
    const long long tid = (long long)blockIdx.x * blockDim.x + threadIdx.x;
    const int seg = (int)(tid >> 5);
    if (seg >= nseg) return;
    const int lp = (int)(tid & 31);
    const float inv = 1.0f / fmaxf(deg[seg], 1.0f);
    float4* p = reinterpret_cast<float4*>(agg + (size_t)seg * D_MODEL + lp * 4);
    float4 v = *p;
    v.x *= inv; v.y *= inv; v.z *= inv; v.w *= inv;
    *p = v;
}

extern "C" void kernel_launch(void* const* d_in, const int* in_sizes, int n_in,
                              void* d_out, int out_size, void* d_ws, size_t ws_size,
                              hipStream_t stream) {
    const float* x  = (const float*)d_in[0];
    const int*   ei = (const int*)d_in[1];
    const int n_inc = in_sizes[1] / 2;
    const int* nodes  = ei;
    const int* hedges = ei + n_inc;

    // params flattened in dict order:
    // v2e: w1 b1 g be w2 b2 | wc | dw1 db1 dg dbe dw2 db2   (P[0..12])
    // e2v: same                                              (P[13..25])
    const float* P[26];
    for (int i = 0; i < 26 && (2 + i) < n_in; ++i) P[i] = (const float*)d_in[2 + i];

    // ---- workspace carve ----
    char*  ws  = (char*)d_ws;
    size_t off = 0;
    auto carve = [&](size_t bytes) -> void* {
        void* p = ws + off;
        off += (bytes + 255) & ~(size_t)255;
        return p;
    };
    _Float16* Wp   = (_Float16*)carve((size_t)10 * WFRAG * sizeof(_Float16));
    float*    bufA = (float*)carve((size_t)N_NODES_C * D_MODEL * sizeof(float));
    float*    bufB = (float*)carve((size_t)N_NODES_C * D_MODEL * sizeof(float));
    float*    degb = (float*)carve((size_t)N_NODES_C * sizeof(float));
    float*    bufC = (float*)carve((size_t)N_HE_C * D_MODEL * sizeof(float));
    int*      minp = (int*)carve(16);
    (void)ws_size; (void)out_size;

    // ---- pack all 10 weight matrices into WMMA fragment order (f16) ----
    const float* wmats[10] = { P[0], P[4], P[6], P[7], P[11],
                               P[13], P[17], P[19], P[20], P[24] };
    for (int i = 0; i < 10; ++i)
        pack_weights_kernel<<<4, 256, 0, stream>>>(wmats[i], Wp + (size_t)i * WFRAG);

    // ---- min(hedges) ----
    hipMemsetAsync(minp, 0x7f, sizeof(int), stream);
    min_reduce_kernel<<<64, 256, 0, stream>>>(hedges, n_inc, minp);

    const unsigned scatter_blocks = (unsigned)(((size_t)n_inc * 32 + 255) / 256);

    // =============== layer 1: v2e (nodes -> hyperedges) ===============
    setconv_mlp_kernel<<<(N_NODES_C + 63) / 64, 128, 0, stream>>>(
        x, bufA, N_NODES_C,
        Wp + 0 * WFRAG, Wp + 1 * WFRAG, Wp + 2 * WFRAG,
        P[1], P[2], P[3], P[5], /*has_wc=*/1);

    hipMemsetAsync(bufB, 0, (size_t)N_HE_C * D_MODEL * sizeof(float), stream);
    hipMemsetAsync(degb, 0, (size_t)N_HE_C * sizeof(float), stream);
    scatter_add_kernel<<<scatter_blocks, 256, 0, stream>>>(
        bufA, nodes, hedges, minp, /*sub_src=*/0, /*sub_tgt=*/1, bufB, degb, n_inc);
    seg_norm_kernel<<<(N_HE_C * 32 + 255) / 256, 256, 0, stream>>>(bufB, degb, N_HE_C);

    setconv_mlp_kernel<<<(N_HE_C + 63) / 64, 128, 0, stream>>>(
        bufB, bufC, N_HE_C,
        Wp + 3 * WFRAG, Wp + 4 * WFRAG, Wp,
        P[8], P[9], P[10], P[12], /*has_wc=*/0);

    // =============== layer 2: e2v (hyperedges -> nodes) ===============
    setconv_mlp_kernel<<<(N_HE_C + 63) / 64, 128, 0, stream>>>(
        bufC, bufA, N_HE_C,
        Wp + 5 * WFRAG, Wp + 6 * WFRAG, Wp + 7 * WFRAG,
        P[14], P[15], P[16], P[18], /*has_wc=*/1);

    hipMemsetAsync(bufB, 0, (size_t)N_NODES_C * D_MODEL * sizeof(float), stream);
    hipMemsetAsync(degb, 0, (size_t)N_NODES_C * sizeof(float), stream);
    scatter_add_kernel<<<scatter_blocks, 256, 0, stream>>>(
        bufA, hedges, nodes, minp, /*sub_src=*/1, /*sub_tgt=*/0, bufB, degb, n_inc);
    seg_norm_kernel<<<((size_t)N_NODES_C * 32 + 255) / 256, 256, 0, stream>>>(
        bufB, degb, N_NODES_C);

    setconv_mlp_kernel<<<(N_NODES_C + 63) / 64, 128, 0, stream>>>(
        bufB, (float*)d_out, N_NODES_C,
        Wp + 8 * WFRAG, Wp + 9 * WFRAG, Wp,
        P[21], P[22], P[23], P[25], /*has_wc=*/0);
}
